// ParaModel_69664369541839
// MI455X (gfx1250) — matrix operand order repaired and checked by reading, not compiled, not run on hardware
//
#include <hip/hip_runtime.h>
#include <stdint.h>
#include <string.h>

#define B_   128
#define L_   256
#define D_   768
#define NEGV (-10000.0f)
#define EPS_ (1e-8f)

typedef __attribute__((ext_vector_type(16))) __bf16 v16bf;
typedef __attribute__((ext_vector_type(8)))  float  v8f;

struct U128 { unsigned x, y, z, w; };
union FragU { U128 u[2]; v16bf v; };

// ---- order-preserving float <-> uint encoding (for atomicMax on floats) ----
__device__ __forceinline__ unsigned fenc(float f) {
  unsigned u = __float_as_uint(f);
  return (u & 0x80000000u) ? ~u : (u | 0x80000000u);
}
__device__ __forceinline__ float fdec(unsigned u) {
  unsigned b = (u & 0x80000000u) ? (u & 0x7fffffffu) : ~u;
  return __uint_as_float(b);
}

// ---- f32 bits -> packed bf16 pair via v_perm_b32 (round-half-up) ----
// result = { bf16(b) , bf16(a) }  (a in low half)
__device__ __forceinline__ unsigned packbf2(unsigned ua, unsigned ub) {
  return __builtin_amdgcn_perm(ub + 0x8000u, ua + 0x8000u, 0x07060302u);
}

// ---------------------------------------------------------------------------
// init: fill row/col max workspace with encoded -3e38
// ---------------------------------------------------------------------------
__global__ void initmax_kernel(unsigned* __restrict__ p, unsigned val, int n) {
  int i = blockIdx.x * blockDim.x + threadIdx.x;
  if (i < n) p[i] = val;
}

// ---------------------------------------------------------------------------
// Phase 1: bf16 WMMA GEMM (v1 . v2^T), fused masked row/col max reduction.
// grid = (4 slabs, B batches), block = 256 (8 waves). Double-buffered LDS,
// register-double-buffered B fragments (partial dscnt waits).
// Wave w: tile-row (w>>1) of the 64-row slab, column half (w&1) -> 8 tiles.
// ---------------------------------------------------------------------------
__global__ __launch_bounds__(256) void gemm_max_kernel(
    const float* __restrict__ v1, const float* __restrict__ mask1,
    const float* __restrict__ v2, const float* __restrict__ mask2,
    unsigned* __restrict__ rowU, unsigned* __restrict__ colU)
{
  constexpr int PAD = 40;                 // bf16 elems per LDS row (80B, conflict-free)
  __shared__ unsigned short sA[2][64 * PAD];
  __shared__ unsigned short sB[2][256 * PAD];
  __shared__ float sM1[64];
  __shared__ float sM2[256];

  const int tid  = threadIdx.x;
  const int slab = blockIdx.x;            // 0..3 (64 rows of L1 each)
  const int b    = blockIdx.y;            // batch

  const int wave  = tid >> 5;
  const int ln    = tid & 31;
  const int trow  = wave >> 1;            // 0..3
  const int chalf = wave & 1;             // 0..1 (128 columns each)
  const int half  = ln >> 4;
  const int lan16 = ln & 15;

  if (tid < 64) sM1[tid] = mask1[b * L_ + slab * 64 + tid];
  sM2[tid] = mask2[b * L_ + tid];

  const float* v1b = v1 + (size_t)(b * L_ + slab * 64) * D_;
  const float* v2b = v2 + (size_t)(b * L_) * D_;

  // per-thread staging coordinates
  const int rA0 = tid >> 3, qA0 = tid & 7;          // A items: tid, tid+256
  const int rA1 = (tid + 256) >> 3, qA1 = tid & 7;

  v8f acc[8];
#pragma unroll
  for (int t = 0; t < 8; ++t)
#pragma unroll
    for (int r = 0; r < 8; ++r) acc[t][r] = 0.0f;

  const int aoff0 = (trow * 16 + lan16) * (PAD * 2) + half * 32;
  const int boff0 = (chalf * 128 + lan16) * (PAD * 2) + half * 32;

  uint4 ra[2], rb[8];

  auto load_chunk = [&](int k0) {
    ra[0] = *(const uint4*)(v1b + (size_t)rA0 * D_ + k0 + qA0 * 4);
    ra[1] = *(const uint4*)(v1b + (size_t)rA1 * D_ + k0 + qA1 * 4);
#pragma unroll
    for (int j = 0; j < 8; ++j) {
      int i = tid + j * 256;
      int row = i >> 3, q = i & 7;
      rb[j] = *(const uint4*)(v2b + (size_t)row * D_ + k0 + q * 4);
    }
  };

  auto stage = [&](int bsel) {
    {
      unsigned* d = (unsigned*)&sA[bsel][rA0 * PAD + qA0 * 4];
      d[0] = packbf2(ra[0].x, ra[0].y);
      d[1] = packbf2(ra[0].z, ra[0].w);
      unsigned* e = (unsigned*)&sA[bsel][rA1 * PAD + qA1 * 4];
      e[0] = packbf2(ra[1].x, ra[1].y);
      e[1] = packbf2(ra[1].z, ra[1].w);
    }
#pragma unroll
    for (int j = 0; j < 8; ++j) {
      int i = tid + j * 256;
      int row = i >> 3, q = i & 7;
      unsigned* d = (unsigned*)&sB[bsel][row * PAD + q * 4];
      d[0] = packbf2(rb[j].x, rb[j].y);
      d[1] = packbf2(rb[j].z, rb[j].w);
    }
  };

  auto compute = [&](int bsel) {
    const char* baseA = (const char*)sA[bsel];
    const char* baseB = (const char*)sB[bsel];
    FragU a;
    a.u[0] = *(const U128*)(baseA + aoff0);
    a.u[1] = *(const U128*)(baseA + aoff0 + 16);

    FragU bf[2];
    // preload tile 0
    bf[0].u[0] = *(const U128*)(baseB + boff0);
    bf[0].u[1] = *(const U128*)(baseB + boff0 + 16);
#pragma unroll
    for (int t = 0; t < 8; ++t) {
      if (t < 7) {  // prefetch next B fragment into alternate regs
        int boff = boff0 + (t + 1) * 16 * (PAD * 2);
        bf[(t + 1) & 1].u[0] = *(const U128*)(baseB + boff);
        bf[(t + 1) & 1].u[1] = *(const U128*)(baseB + boff + 16);
      }
      acc[t] = __builtin_amdgcn_wmma_f32_16x16x32_bf16(
          false, a.v, false, bf[t & 1].v, (short)0, acc[t], false, false);
    }
  };

  // ---- software-pipelined K loop (double-buffered LDS) ----
  load_chunk(0);
  stage(0);
  __syncthreads();
  int bsel = 0;
  for (int k0 = 0; k0 < D_; k0 += 32) {
    const bool has_next = (k0 + 32 < D_);
    if (has_next) load_chunk(k0 + 32);   // global loads in flight during compute
    compute(bsel);
    if (has_next) stage(bsel ^ 1);       // waits loads, converts, ds_stores
    __syncthreads();
    bsel ^= 1;
  }

  // ---- fused masked max reductions ----
  float rmv[8];
#pragma unroll
  for (int r = 0; r < 8; ++r) rmv[r] = sM1[trow * 16 + 8 * half + r];

  float rowc[8];
#pragma unroll
  for (int r = 0; r < 8; ++r) rowc[r] = -3.0e38f;

#pragma unroll
  for (int t = 0; t < 8; ++t) {
    int m = chalf * 128 + t * 16 + lan16;
    float cm = sM2[m];
    float colc = -3.0e38f;
#pragma unroll
    for (int r = 0; r < 8; ++r) {
      float val = (rmv[r] > 0.0f && cm > 0.0f) ? acc[t][r] : NEGV;
      rowc[r] = fmaxf(rowc[r], val);
      colc    = fmaxf(colc, val);
    }
    colc = fmaxf(colc, __shfl_xor(colc, 16, 32));   // combine M halves
    if (ln < 16) atomicMax(&colU[b * L_ + m], fenc(colc));
  }

#pragma unroll
  for (int r = 0; r < 8; ++r) {
    float v = rowc[r];
    v = fmaxf(v, __shfl_xor(v, 1, 32));             // reduce over N (16 lanes)
    v = fmaxf(v, __shfl_xor(v, 2, 32));
    v = fmaxf(v, __shfl_xor(v, 4, 32));
    v = fmaxf(v, __shfl_xor(v, 8, 32));
    if (lan16 == 0)
      atomicMax(&rowU[b * L_ + slab * 64 + trow * 16 + 8 * half + r], fenc(v));
  }
}

// ---------------------------------------------------------------------------
// Phase 2a: masked softmax over L for both sides. grid (B, 2), block 256.
// ---------------------------------------------------------------------------
__global__ __launch_bounds__(256) void softmax_kernel(
    const float* __restrict__ mask1, const float* __restrict__ mask2,
    const unsigned* __restrict__ rowU, const unsigned* __restrict__ colU,
    float* __restrict__ attn1, float* __restrict__ attn2)
{
  __shared__ float red[256];
  const int tid = threadIdx.x, b = blockIdx.x, side = blockIdx.y;

  const float*    mask = side ? mask2 : mask1;
  const unsigned* U    = side ? colU  : rowU;
  float*          attn = side ? attn2 : attn1;

  float mk = mask[b * L_ + tid];
  float s  = (mk > 0.0f) ? (-fdec(U[b * L_ + tid]) * 0.01f) : NEGV;

  red[tid] = s; __syncthreads();
  for (int o = 128; o > 0; o >>= 1) { if (tid < o) red[tid] = fmaxf(red[tid], red[tid + o]); __syncthreads(); }
  float mx = red[0]; __syncthreads();
  float e = __expf(s - mx);
  red[tid] = e; __syncthreads();
  for (int o = 128; o > 0; o >>= 1) { if (tid < o) red[tid] += red[tid + o]; __syncthreads(); }
  float sum = red[0];

  attn[b * L_ + tid] = e / sum;
}

// ---------------------------------------------------------------------------
// Phase 2b: attention pooling p = sum_l attn[l] * v[b,l,:].
// grid (B, 6): y/3 = side, y%3 = 256-dim chunk. block 256.
// ---------------------------------------------------------------------------
__global__ __launch_bounds__(256) void pool_kernel(
    const float* __restrict__ v1, const float* __restrict__ v2,
    const float* __restrict__ attn1, const float* __restrict__ attn2,
    float* __restrict__ p1, float* __restrict__ p2)
{
  __shared__ float at[256];
  const int tid = threadIdx.x, b = blockIdx.x;
  const int side  = blockIdx.y / 3;
  const int chunk = blockIdx.y % 3;

  const float* v    = side ? v2    : v1;
  const float* attn = side ? attn2 : attn1;
  float*       p    = side ? p2    : p1;

  at[tid] = attn[b * L_ + tid];
  __syncthreads();

  const int d = chunk * 256 + tid;
  float acc = 0.0f;
  for (int l = 0; l < L_; ++l)
    acc = fmaf(at[l], v[(size_t)(b * L_ + l) * D_ + d], acc);
  p[b * D_ + d] = acc;
}

// ---------------------------------------------------------------------------
// Phase 2c: cosine similarity. grid B, block 256.
// ---------------------------------------------------------------------------
__global__ __launch_bounds__(256) void cosine_kernel(
    const float* __restrict__ p1, const float* __restrict__ p2,
    float* __restrict__ out)
{
  __shared__ float red[256];
  const int tid = threadIdx.x, b = blockIdx.x;

  float dp = 0.0f, n1 = 0.0f, n2 = 0.0f;
#pragma unroll
  for (int j = 0; j < 3; ++j) {
    int d = tid + j * 256;
    float x = p1[b * D_ + d], y = p2[b * D_ + d];
    dp = fmaf(x, y, dp);
    n1 = fmaf(x, x, n1);
    n2 = fmaf(y, y, n2);
  }
  red[tid] = dp; __syncthreads();
  for (int o = 128; o > 0; o >>= 1) { if (tid < o) red[tid] += red[tid + o]; __syncthreads(); }
  dp = red[0]; __syncthreads();
  red[tid] = n1; __syncthreads();
  for (int o = 128; o > 0; o >>= 1) { if (tid < o) red[tid] += red[tid + o]; __syncthreads(); }
  n1 = red[0]; __syncthreads();
  red[tid] = n2; __syncthreads();
  for (int o = 128; o > 0; o >>= 1) { if (tid < o) red[tid] += red[tid + o]; __syncthreads(); }
  n2 = red[0];

  if (tid == 0)
    out[b] = dp / (fmaxf(sqrtf(n1), EPS_) * fmaxf(sqrtf(n2), EPS_));
}

// ---------------------------------------------------------------------------
extern "C" void kernel_launch(void* const* d_in, const int* in_sizes, int n_in,
                              void* d_out, int out_size, void* d_ws, size_t ws_size,
                              hipStream_t stream) {
  const float* v1    = (const float*)d_in[0];
  const float* mask1 = (const float*)d_in[1];
  const float* v2    = (const float*)d_in[2];
  const float* mask2 = (const float*)d_in[3];
  float* out = (float*)d_out;

  // workspace layout
  unsigned* rowU  = (unsigned*)d_ws;           // B*L
  unsigned* colU  = rowU + B_ * L_;            // B*L
  float*    attn1 = (float*)(colU + B_ * L_);  // B*L
  float*    attn2 = attn1 + B_ * L_;           // B*L
  float*    p1    = attn2 + B_ * L_;           // B*D
  float*    p2    = p1 + B_ * D_;              // B*D

  // host-side order-preserving encode of -3e38f
  float init_f = -3.0e38f;
  unsigned u; memcpy(&u, &init_f, 4);
  unsigned initv = (u & 0x80000000u) ? ~u : (u | 0x80000000u);

  int ninit = 2 * B_ * L_;
  initmax_kernel<<<(ninit + 255) / 256, 256, 0, stream>>>(rowU, initv, ninit);
  gemm_max_kernel<<<dim3(4, B_), 256, 0, stream>>>(v1, mask1, v2, mask2, rowU, colU);
  softmax_kernel<<<dim3(B_, 2), 256, 0, stream>>>(mask1, mask2, rowU, colU, attn1, attn2);
  pool_kernel<<<dim3(B_, 6), 256, 0, stream>>>(v1, v2, attn1, attn2, p1, p2);
  cosine_kernel<<<B_, 256, 0, stream>>>(p1, p2, out);
}
